// MutiHeadSelfAttention_62328565400288
// MI455X (gfx1250) — matrix operand order repaired and checked by reading, not compiled, not run on hardware
//
#include <hip/hip_runtime.h>
#include <hip/hip_fp16.h>

// ---------------------------------------------------------------------------
// MHA + RoPE for MI455X (gfx1250, wave32, WMMA 16x16x32 f16, f32 accum).
// B=2, S=2048, D=1024, H=16, DK=64.
//
// Workspace (f16 elements), M1 = 1024*1024:
//   [0,1M) WQh [1M,2M) WKh [2M,3M) WVh [3M,4M) WOh
//   [4M,8M) Xh   [8M,12M) Qh [B,H,S,DK]   [12M,16M) Kh [B,H,S,DK]
//   [16M,20M) Vt [B,H,DK,S]   [20M,24M) Ch [B,S,D]      (48 MB total)
// ---------------------------------------------------------------------------

typedef _Float16 f16;
typedef _Float16 v16h __attribute__((ext_vector_type(16)));
typedef float    v8f  __attribute__((ext_vector_type(8)));
typedef unsigned int v4u __attribute__((ext_vector_type(4)));
typedef int i4v __attribute__((vector_size(16)));  // builtin's pointee type

union H16 { v16h h; v4u q[2]; f16 e[16]; };

#define DEV static __device__ __forceinline__

DEV v8f wmma32(v16h a, v16h b, v8f c) {
  return __builtin_amdgcn_wmma_f32_16x16x32_f16(false, a, false, b, (short)0, c,
                                                false, false);
}
DEV v8f vzero8() { v8f z = {0.f,0.f,0.f,0.f,0.f,0.f,0.f,0.f}; return z; }

// A-tile (16x32 f16): per-lane halves at +0 and +16
DEV void loadA(H16& d, const f16* p) {
  d.q[0] = *(const v4u*)(p);
  d.q[1] = *(const v4u*)(p + 16);
}
// B-tile (32x16 f16): per-lane halves at +0 and +16
DEV void loadB(H16& d, const f16* p) {
  d.q[0] = *(const v4u*)(p);
  d.q[1] = *(const v4u*)(p + 16);
}

enum { SEQ = 2048, DM = 1024, NH = 16, DK = 64, BB = 2 };
enum { KST = 1032 };  // LDS A row stride in halves (2064B: aligned, no bank hit)

// ---------------------------------------------------------------------------
// Async global->LDS copy (CDNA5 GLOBAL_LOAD_ASYNC_TO_LDS_B128, ASYNCcnt),
// with a portable fallback through VGPRs.
// Builtin signature (from compiler diagnostic): param0 = addrspace(1) i4v*,
// param1 = addrspace(3) i4v*, then imm offset, imm cpol.
// ---------------------------------------------------------------------------
#if __has_builtin(__builtin_amdgcn_global_load_async_to_lds_b128)
#define HAVE_ASYNC_LDS 1
#else
#define HAVE_ASYNC_LDS 0
#endif

DEV void cp_b128_g2l(const f16* g, f16* l) {
#if HAVE_ASYNC_LDS
  __builtin_amdgcn_global_load_async_to_lds_b128(
      (__attribute__((address_space(1))) i4v*)(i4v*)(void*)(f16*)g,
      (__attribute__((address_space(3))) i4v*)(i4v*)(void*)l, 0, 0);
#else
  *(v4u*)l = *(const v4u*)g;
#endif
}

DEV void async_fence() {
#if HAVE_ASYNC_LDS
#if __has_builtin(__builtin_amdgcn_s_wait_asynccnt)
  __builtin_amdgcn_s_wait_asynccnt(0);
#else
  asm volatile("s_wait_asynccnt 0x0" ::: "memory");
#endif
#endif
}

// stage full 32 x 1024 f16 panel into LDS once per block (256 threads)
DEV void stage_panel(f16* As, const f16* src, int tid) {
#pragma unroll
  for (int j = 0; j < 16; ++j) {
    int idx = tid * 128 + j * 8;  // 32768 halves total
    int row = idx >> 10, col = idx & (DM - 1);
    cp_b128_g2l(&src[row * DM + col], &As[row * KST + col]);
  }
  async_fence();
}

// ---------------------------------------------------------------------------
__global__ void cvt_f32_to_f16(const float* __restrict__ src,
                               f16* __restrict__ dst, int n) {
  int i = blockIdx.x * blockDim.x + threadIdx.x;
  if (i < n) dst[i] = (f16)src[i];
}

// ---------------------------------------------------------------------------
// Projection GEMM core: one wave = 32(M) x 64(N); copy-free ping-pong over
// the full K=1024 (A resident in LDS). 8 WMMA per 32-K stage.
// ---------------------------------------------------------------------------
struct ProjCtx {
  const f16* arow[2];   // LDS A row base per row-tile
  const f16* brow[4];   // global B row base per col-tile
};

DEV void proj_k_pass(const ProjCtx& c, v8f acc[2][4]) {
  H16 A0[2], A1[2], B0[4], B1[4];
#pragma unroll
  for (int rt = 0; rt < 2; ++rt) loadA(A0[rt], c.arow[rt]);
#pragma unroll
  for (int u = 0; u < 4; ++u) loadB(B0[u], c.brow[u]);

#pragma unroll 1
  for (int k = 0; k < DM; k += 64) {
#pragma unroll
    for (int rt = 0; rt < 2; ++rt) loadA(A1[rt], c.arow[rt] + k + 32);
#pragma unroll
    for (int u = 0; u < 4; ++u) loadB(B1[u], c.brow[u] + k + 32);
#pragma unroll
    for (int rt = 0; rt < 2; ++rt)
#pragma unroll
      for (int u = 0; u < 4; ++u)
        acc[rt][u] = wmma32(A0[rt].h, B0[u].h, acc[rt][u]);

    const int kn = (k + 64) & (DM - 1);  // wrap: harmless reload on last
#pragma unroll
    for (int rt = 0; rt < 2; ++rt) loadA(A0[rt], c.arow[rt] + kn);
#pragma unroll
    for (int u = 0; u < 4; ++u) loadB(B0[u], c.brow[u] + kn);
#pragma unroll
    for (int rt = 0; rt < 2; ++rt)
#pragma unroll
      for (int u = 0; u < 4; ++u)
        acc[rt][u] = wmma32(A1[rt].h, B1[u].h, acc[rt][u]);
  }
}

// ---------------------------------------------------------------------------
// Kernel 1: fused QKV projection + RoPE.
// grid = (128 m-tiles of 32 rows, 3 mats), block = 256 (8 waves).
// ---------------------------------------------------------------------------
__global__ __launch_bounds__(256) void qkv_rope_kernel(
    const f16* __restrict__ xh, const f16* __restrict__ wh,
    f16* __restrict__ qh, f16* __restrict__ kh, f16* __restrict__ vt,
    const int* __restrict__ tpos) {
  __shared__ __attribute__((aligned(16))) f16 As[32 * KST];  // 66 KB of 320 KB

  const int tid = threadIdx.x;
  const int r0  = blockIdx.x * 32;
  const int mat = blockIdx.y;  // 0=Q 1=K 2=V

  stage_panel(As, xh + (size_t)r0 * DM, tid);
  __syncthreads();  // panel is read-only from here: no more barriers

  const int lane = tid & 31, wv = tid >> 5;
  const int lrow = lane & 15, lhi = lane >> 4;
  const f16* wbase = wh + (size_t)mat * (DM * DM);

  ProjCtx c;
#pragma unroll
  for (int rt = 0; rt < 2; ++rt)
    c.arow[rt] = As + (16 * rt + lrow) * KST + (lhi ? 8 : 0);

  for (int nc = 0; nc < 2; ++nc) {
    const int n0 = nc * 512 + wv * 64;  // 64-aligned: one head per pass
#pragma unroll
    for (int u = 0; u < 4; ++u)
      c.brow[u] = wbase + (size_t)(n0 + 16 * u + lrow) * DM + (lhi ? 8 : 0);

    v8f acc[2][4];
#pragma unroll
    for (int rt = 0; rt < 2; ++rt)
#pragma unroll
      for (int u = 0; u < 4; ++u) acc[rt][u] = vzero8();

    proj_k_pass(c, acc);

    const int hd = n0 >> 6;
    if (mat < 2) {
      f16* dst = (mat == 0) ? qh : kh;
#pragma unroll
      for (int rt = 0; rt < 2; ++rt)
#pragma unroll
        for (int u = 0; u < 4; ++u) {
          const int dk     = 16 * u + lrow;
          const int parity = dk & 1;
          // theta^(-2j/dk) = 2^(-j * 2/64 * log2(10000))
          const float invf =
              exp2f(-(float)(dk >> 1) * 0.4152410118609203f);
#pragma unroll
          for (int v = 0; v < 8; ++v) {
            int r = r0 + 16 * rt + v + lhi * 8;
            int b = r >> 11, s = r & (SEQ - 1);
            float pos  = (float)tpos[s];
            float self = acc[rt][u][v];
            float part = __shfl_xor(self, 1, 32);
            float sn, cs;
            __sincosf(pos * invf, &sn, &cs);
            float val = parity ? (part * sn + self * cs)
                               : (self * cs - part * sn);
            dst[(((size_t)(b * NH + hd) * SEQ + s) * DK) + dk] = (f16)val;
          }
        }
    } else {
#pragma unroll
      for (int rt = 0; rt < 2; ++rt) {
        int rb = r0 + 16 * rt;
        int b = rb >> 11, s0 = rb & (SEQ - 1);
#pragma unroll
        for (int u = 0; u < 4; ++u) {
          const int dk = 16 * u + lrow;
          H16 pk;
#pragma unroll
          for (int v = 0; v < 8; ++v) pk.e[v] = (f16)acc[rt][u][v];
          f16* dst =
              vt + ((size_t)(b * NH + hd) * DK + dk) * SEQ + s0 + lhi * 8;
          *(v4u*)dst = pk.q[0];
        }
      }
    }
  }
}

// ---------------------------------------------------------------------------
// Kernel 2: causal flash attention. grid=(B*H, S/128), block=128 (4 waves).
// Wave = 32 Q rows (2 row-tiles) sharing K/V tiles; base-2 online softmax.
// ---------------------------------------------------------------------------
__global__ __launch_bounds__(128) void attn_kernel(
    const f16* __restrict__ qh, const f16* __restrict__ kh,
    const f16* __restrict__ vt, f16* __restrict__ ch) {
  __shared__ __attribute__((aligned(16))) f16 Ps[4][32 * 40];

  const int bh = blockIdx.x, b = bh >> 4, h = bh & (NH - 1);
  const int tid = threadIdx.x, wave = tid >> 5, lane = tid & 31;
  const int lrow = lane & 15, lhi = lane >> 4;
  const int q0 = blockIdx.y * 128 + wave * 32;

  H16 AQ[2][2];
#pragma unroll
  for (int rt = 0; rt < 2; ++rt) {
    const f16* qrow =
        qh + ((size_t)bh * SEQ + q0 + 16 * rt + lrow) * DK + lhi * 8;
    loadA(AQ[rt][0], qrow);
    loadA(AQ[rt][1], qrow + 32);
  }

  v8f oa[2][4];
  float mr[2][8], lr[2][8];
#pragma unroll
  for (int rt = 0; rt < 2; ++rt) {
#pragma unroll
    for (int u = 0; u < 4; ++u) oa[rt][u] = vzero8();
#pragma unroll
    for (int v = 0; v < 8; ++v) { mr[rt][v] = -1e30f; lr[rt][v] = 0.f; }
  }

  const float SCL2 = 0.18033688011112042f;  // (1/sqrt(64)) * log2(e)

  f16* pb = &Ps[wave][0];
  const int kend = q0 + 32;
#pragma unroll 1
  for (int k0 = 0; k0 < kend; k0 += 32) {
    H16 BK[2][2];
#pragma unroll
    for (int t = 0; t < 2; ++t) {
      const f16* krow =
          kh + ((size_t)bh * SEQ + k0 + 16 * t + lrow) * DK + lhi * 8;
      loadB(BK[t][0], krow);
      loadB(BK[t][1], krow + 32);
    }
    // V tiles issued early: in flight across scores + softmax + LDS trip
    H16 BV[4];
#pragma unroll
    for (int u = 0; u < 4; ++u) {
      const f16* vrow =
          vt + ((size_t)bh * DK + 16 * u + lrow) * SEQ + k0 + lhi * 8;
      loadB(BV[u], vrow);
    }

    v8f sc[2][2];
#pragma unroll
    for (int rt = 0; rt < 2; ++rt)
#pragma unroll
      for (int t = 0; t < 2; ++t) {
        v8f z = vzero8();
        z = wmma32(AQ[rt][0].h, BK[t][0].h, z);
        z = wmma32(AQ[rt][1].h, BK[t][1].h, z);
        sc[rt][t] = z;
      }

#pragma unroll
    for (int rt = 0; rt < 2; ++rt) {
      float p[2][8];
#pragma unroll
      for (int t = 0; t < 2; ++t)
#pragma unroll
        for (int v = 0; v < 8; ++v) {
          float s = sc[rt][t][v] * SCL2;
          int key = k0 + 16 * t + lrow;
          int q   = q0 + 16 * rt + v + lhi * 8;
          p[t][v] = (key <= q) ? s : -1e30f;
        }
#pragma unroll
      for (int v = 0; v < 8; ++v) {
        float mx = fmaxf(p[0][v], p[1][v]);
        mx = fmaxf(mx, __shfl_xor(mx, 1, 32));
        mx = fmaxf(mx, __shfl_xor(mx, 2, 32));
        mx = fmaxf(mx, __shfl_xor(mx, 4, 32));
        mx = fmaxf(mx, __shfl_xor(mx, 8, 32));
        float nm   = fmaxf(mr[rt][v], mx);
        float corr = exp2f(mr[rt][v] - nm);  // native v_exp_f32
        mr[rt][v] = nm;
        float e0 = exp2f(p[0][v] - nm), e1 = exp2f(p[1][v] - nm);
        p[0][v] = e0; p[1][v] = e1;
        float rs = e0 + e1;
        rs += __shfl_xor(rs, 1, 32);
        rs += __shfl_xor(rs, 2, 32);
        rs += __shfl_xor(rs, 4, 32);
        rs += __shfl_xor(rs, 8, 32);
        lr[rt][v] = lr[rt][v] * corr + rs;
#pragma unroll
        for (int u = 0; u < 4; ++u) oa[rt][u][v] *= corr;
      }
#pragma unroll
      for (int t = 0; t < 2; ++t)
#pragma unroll
        for (int v = 0; v < 8; ++v) {
          int row = 16 * rt + v + lhi * 8, col = 16 * t + lrow;
          pb[row * 40 + col] = (f16)p[t][v];
        }
    }

    asm volatile("s_wait_dscnt 0x0" ::: "memory");

#pragma unroll
    for (int rt = 0; rt < 2; ++rt) {
      H16 AP;
      const f16* par = pb + (16 * rt + lrow) * 40 + lhi * 8;
      AP.q[0] = *(const v4u*)par;
      AP.q[1] = *(const v4u*)(par + 16);
#pragma unroll
      for (int u = 0; u < 4; ++u)
        oa[rt][u] = wmma32(AP.h, BV[u].h, oa[rt][u]);
    }
  }

  // one v_rcp per row instead of 4 divides
#pragma unroll
  for (int rt = 0; rt < 2; ++rt)
#pragma unroll
    for (int v = 0; v < 8; ++v) {
      float inv = __builtin_amdgcn_rcpf(lr[rt][v]);
      int q = q0 + 16 * rt + v + lhi * 8;
#pragma unroll
      for (int u = 0; u < 4; ++u) {
        int dk = 16 * u + lrow;
        ch[((size_t)b * SEQ + q) * DM + h * DK + dk] =
            (f16)(oa[rt][u][v] * inv);
      }
    }
}

// ---------------------------------------------------------------------------
// Kernel 3: output projection (f32 out). Same 32x64-per-wave blocking.
// ---------------------------------------------------------------------------
__global__ __launch_bounds__(256) void oproj_kernel(
    const f16* __restrict__ ch, const f16* __restrict__ woh,
    float* __restrict__ out) {
  __shared__ __attribute__((aligned(16))) f16 As[32 * KST];

  const int tid = threadIdx.x;
  const int r0  = blockIdx.x * 32;

  stage_panel(As, ch + (size_t)r0 * DM, tid);
  __syncthreads();

  const int lane = tid & 31, wv = tid >> 5;
  const int lrow = lane & 15, lhi = lane >> 4;

  ProjCtx c;
#pragma unroll
  for (int rt = 0; rt < 2; ++rt)
    c.arow[rt] = As + (16 * rt + lrow) * KST + (lhi ? 8 : 0);

  for (int nc = 0; nc < 2; ++nc) {
    const int n0 = nc * 512 + wv * 64;
#pragma unroll
    for (int u = 0; u < 4; ++u)
      c.brow[u] = woh + (size_t)(n0 + 16 * u + lrow) * DM + (lhi ? 8 : 0);

    v8f acc[2][4];
#pragma unroll
    for (int rt = 0; rt < 2; ++rt)
#pragma unroll
      for (int u = 0; u < 4; ++u) acc[rt][u] = vzero8();

    proj_k_pass(c, acc);

#pragma unroll
    for (int rt = 0; rt < 2; ++rt)
#pragma unroll
      for (int u = 0; u < 4; ++u)
#pragma unroll
        for (int v = 0; v < 8; ++v) {
          int r = r0 + 16 * rt + v + lhi * 8;
          out[(size_t)r * DM + (n0 + 16 * u + lrow)] = acc[rt][u][v];
        }
  }
}

// ---------------------------------------------------------------------------
extern "C" void kernel_launch(void* const* d_in, const int* in_sizes, int n_in,
                              void* d_out, int out_size, void* d_ws,
                              size_t ws_size, hipStream_t stream) {
  const float* x    = (const float*)d_in[0];
  const int*   tpos = (const int*)d_in[1];
  const float* Wq   = (const float*)d_in[2];
  const float* Wk   = (const float*)d_in[3];
  const float* Wv   = (const float*)d_in[4];
  const float* Wo   = (const float*)d_in[5];
  float* out = (float*)d_out;

  const size_t M1 = (size_t)DM * DM;
  f16* ws  = (f16*)d_ws;
  f16* WQh = ws;
  f16* WOh = ws + 3 * M1;
  f16* Xh  = ws + 4 * M1;
  f16* Qh  = ws + 8 * M1;
  f16* Kh  = ws + 12 * M1;
  f16* Vt  = ws + 16 * M1;
  f16* Ch  = ws + 20 * M1;

  const int nW = (int)M1;
  const int nX = BB * SEQ * DM;

  cvt_f32_to_f16<<<(nW + 255) / 256, 256, 0, stream>>>(Wq, WQh, nW);
  cvt_f32_to_f16<<<(nW + 255) / 256, 256, 0, stream>>>(Wk, ws + 1 * M1, nW);
  cvt_f32_to_f16<<<(nW + 255) / 256, 256, 0, stream>>>(Wv, ws + 2 * M1, nW);
  cvt_f32_to_f16<<<(nW + 255) / 256, 256, 0, stream>>>(Wo, WOh, nW);
  cvt_f32_to_f16<<<(nX + 255) / 256, 256, 0, stream>>>(x, Xh, nX);

  qkv_rope_kernel<<<dim3((BB * SEQ) / 32, 3), 256, 0, stream>>>(
      Xh, WQh, Qh, Kh, Vt, tpos);

  attn_kernel<<<dim3(BB * NH, SEQ / 128), 128, 0, stream>>>(Qh, Kh, Vt, Ch);

  oproj_kernel<<<(BB * SEQ) / 32, 256, 0, stream>>>(Ch, WOh, out);
}
